// RNN_21552145891633
// MI455X (gfx1250) — matrix operand order repaired and checked by reading, not compiled, not run on hardware
//
#include <hip/hip_runtime.h>
#include <hip/hip_bf16.h>

// Problem constants (match the reference)
#define BB 512
#define TT 512
#define DD 64
#define HH 512

typedef __attribute__((ext_vector_type(16))) __bf16 v16bf;
typedef __attribute__((ext_vector_type(8)))  __bf16 v8bf;
typedef __attribute__((ext_vector_type(8)))  float  v8f;
typedef __attribute__((ext_vector_type(4)))  float  f32x4;

// ---------------------------------------------------------------------------
// Setup: zero h0 + barrier counter, transpose-convert W_x [D,H] and W_h [H,H]
// (f32, row-major) into bf16 WT_x [H,D], WT_h [H,H] so that B-matrix WMMA
// fragments are two contiguous 16B runs per lane.
// ---------------------------------------------------------------------------
__global__ void rnn_setup_kernel(const float* __restrict__ Wx,
                                 const float* __restrict__ Wh,
                                 __bf16* __restrict__ WTx,
                                 __bf16* __restrict__ WTh,
                                 __bf16* __restrict__ h0,
                                 unsigned int* __restrict__ bar) {
    const int i = blockIdx.x * blockDim.x + threadIdx.x; // 0 .. H*H-1
    if (i == 0) *bar = 0u;
    if (i < HH * HH) {
        // Wh[k][n] -> WTh[n][k]
        const int k = i / HH, n = i % HH;
        WTh[(size_t)n * HH + k] = (__bf16)Wh[i];
        h0[i] = (__bf16)0.0f;
    }
    if (i < DD * HH) {
        // Wx[d][n] -> WTx[n][d]
        const int d = i / HH, n = i % HH;
        WTx[(size_t)n * DD + d] = (__bf16)Wx[i];
    }
}

// ---------------------------------------------------------------------------
// Persistent recurrence kernel.
// Grid: 64 workgroups (8x8 tiling of 512x512 h), 128 threads (4 waves).
// Wave w owns rows [tileM + 16w, +16); 4 accumulators cover 64 columns.
// W_h slice (64 cols x 512 K, bf16) lives in LDS, XOR-swizzled, 64 KB.
// W_x fragments (t-invariant) are held in registers for the whole run.
// One device-scope barrier per step; h ping-pongs in one contiguous buffer.
// ---------------------------------------------------------------------------
__global__ void __launch_bounds__(128, 1)
rnn_step_kernel(const float* __restrict__ x,     // [B,T,D] f32
                const float* __restrict__ brnn,  // [H]
                const __bf16* __restrict__ WTx,  // [H,D]
                const __bf16* __restrict__ WTh,  // [H,H]
                __bf16* __restrict__ hbase,      // [2,B,H] ping-pong
                unsigned int* __restrict__ bar) {
    // LDS: 64 rows (local n) x 512 K bf16, stored as 16B groups with XOR
    // swizzle: group g of row nl at position (g ^ (nl & 15)).
    __shared__ __bf16 ldsW[64 * 512];

    const int l     = threadIdx.x & 31;        // lane
    const int w     = threadIdx.x >> 5;        // wave 0..3
    const int tileM = (blockIdx.x >> 3) * 64;
    const int tileN = (blockIdx.x & 7) * 64;

    const int kSelA = (l < 16) ? 0 : 8;        // A-frag K base within chunk
    const int kbB   = (l < 16) ? 0 : 1;        // B-frag group base within chunk
    const int rA    = tileM + w * 16 + (l & 15);            // A row (batch idx)
    const int mOut  = tileM + w * 16 + ((l < 16) ? 0 : 8);  // C/D row base
    const int nCol  = (l & 15);                             // C/D col & swizzle key

    // ---- stage WTh slice for columns [tileN, tileN+64) into LDS, swizzled ----
    for (int gi = threadIdx.x; gi < 64 * 64; gi += 128) {
        const int nl = gi >> 6;        // local column 0..63
        const int g  = gi & 63;        // 16B group (8 bf16) within row
        v8bf v = *(const v8bf*)(WTh + (size_t)(tileN + nl) * HH + g * 8);
        *(v8bf*)(ldsW + nl * 512 + ((g ^ (nl & 15)) << 3)) = v;
    }
    __syncthreads();

    // ---- hoist the t-invariant W_x B-fragments into registers ----
    v16bf wxF[2][4];
#pragma unroll
    for (int kc = 0; kc < 2; ++kc)
#pragma unroll
        for (int s = 0; s < 4; ++s) {
            const __bf16* wp = WTx + (size_t)(tileN + s * 16 + nCol) * DD
                               + kc * 32 + (kbB << 3);
            v8bf b0 = *(const v8bf*)(wp);
            v8bf b1 = *(const v8bf*)(wp + 16);
            wxF[kc][s] = __builtin_shufflevector(b0, b1,
                0,1,2,3,4,5,6,7,8,9,10,11,12,13,14,15);
        }

    // Per-lane bias for each 16-col subtile (C/D col = l&15)
    float bias[4];
#pragma unroll
    for (int s = 0; s < 4; ++s) bias[s] = brnn[tileN + s * 16 + nCol];

    // LDS row base per subtile (row nl = s*16 + nCol; swizzle key = nCol)
    const __bf16* ldsRow[4];
#pragma unroll
    for (int s = 0; s < 4; ++s) ldsRow[s] = ldsW + (s * 16 + nCol) * 512;

#pragma unroll 1
    for (int t = 0; t < TT; ++t) {
        const __bf16* __restrict__ hprev = hbase + (size_t)(t & 1) * (BB * HH);
        __bf16* __restrict__ hnext       = hbase + (size_t)((t + 1) & 1) * (BB * HH);

        v8f acc[4];
#pragma unroll
        for (int s = 0; s < 4; ++s)
#pragma unroll
            for (int r = 0; r < 8; ++r) acc[s][r] = bias[s];

        // ---- K-chunks 0..1: input projection x_t @ W_x (D = 64 = 2x32) ----
#pragma unroll
        for (int kc = 0; kc < 2; ++kc) {
            const float* xp = x + (size_t)rA * (TT * DD) + (size_t)t * DD
                              + kc * 32 + kSelA;
            f32x4 f0 = *(const f32x4*)(xp + 0);
            f32x4 f1 = *(const f32x4*)(xp + 4);
            f32x4 f2 = *(const f32x4*)(xp + 16);
            f32x4 f3 = *(const f32x4*)(xp + 20);
            v16bf aFrag;
            aFrag[0]=(__bf16)f0.x; aFrag[1]=(__bf16)f0.y; aFrag[2]=(__bf16)f0.z; aFrag[3]=(__bf16)f0.w;
            aFrag[4]=(__bf16)f1.x; aFrag[5]=(__bf16)f1.y; aFrag[6]=(__bf16)f1.z; aFrag[7]=(__bf16)f1.w;
            aFrag[8]=(__bf16)f2.x; aFrag[9]=(__bf16)f2.y; aFrag[10]=(__bf16)f2.z; aFrag[11]=(__bf16)f2.w;
            aFrag[12]=(__bf16)f3.x; aFrag[13]=(__bf16)f3.y; aFrag[14]=(__bf16)f3.z; aFrag[15]=(__bf16)f3.w;
#pragma unroll
            for (int s = 0; s < 4; ++s) {
                acc[s] = __builtin_amdgcn_wmma_f32_16x16x32_bf16(
                    false, aFrag, false, wxF[kc][s], (short)0, acc[s], false, false);
            }
        }

        // ---- K-chunks 2..17: recurrent part h_{t-1} @ W_h (H = 512 = 16x32),
        //      B fragments from swizzled LDS ----
#pragma unroll 2
        for (int kc = 0; kc < 16; ++kc) {
            const __bf16* hp = hprev + (size_t)rA * HH + kc * 32 + kSelA;
            v8bf a0 = *(const v8bf*)(hp);
            v8bf a1 = *(const v8bf*)(hp + 16);
            v16bf aFrag = __builtin_shufflevector(a0, a1,
                0,1,2,3,4,5,6,7,8,9,10,11,12,13,14,15);
            const int g0 = kc * 4 + kbB;  // first 16B group of this fragment
#pragma unroll
            for (int s = 0; s < 4; ++s) {
                v8bf b0 = *(const v8bf*)(ldsRow[s] + (((g0 + 0) ^ nCol) << 3));
                v8bf b1 = *(const v8bf*)(ldsRow[s] + (((g0 + 2) ^ nCol) << 3));
                v16bf bFrag = __builtin_shufflevector(b0, b1,
                    0,1,2,3,4,5,6,7,8,9,10,11,12,13,14,15);
                acc[s] = __builtin_amdgcn_wmma_f32_16x16x32_bf16(
                    false, aFrag, false, bFrag, (short)0, acc[s], false, false);
            }
        }

        // ---- relu + store h_t (bf16) ----
#pragma unroll
        for (int s = 0; s < 4; ++s) {
            const int n = tileN + s * 16 + nCol;
#pragma unroll
            for (int r = 0; r < 8; ++r) {
                float v = acc[s][r];
                v = v > 0.0f ? v : 0.0f;
                hnext[(size_t)(mOut + r) * HH + n] = (__bf16)v;
            }
        }

        // ---- prefetch next step's x row while we wait at the barrier ----
        if (t + 1 < TT) {
            __builtin_prefetch(x + (size_t)rA * (TT * DD) + (size_t)(t + 1) * DD, 0, 3);
        }

        // ---- device-scope barrier between steps ----
        __threadfence();      // release: make our h_t stores device-visible
        __syncthreads();
        if (threadIdx.x == 0) {
            __hip_atomic_fetch_add(bar, 1u, __ATOMIC_RELEASE, __HIP_MEMORY_SCOPE_AGENT);
            const unsigned int target = (unsigned int)(t + 1) * gridDim.x;
            while (__hip_atomic_load(bar, __ATOMIC_ACQUIRE, __HIP_MEMORY_SCOPE_AGENT) < target) {
                __builtin_amdgcn_s_sleep(8);
            }
        }
        __syncthreads();
        __threadfence();      // acquire: invalidate so next step sees other WGPs' h_t
    }
}

// ---------------------------------------------------------------------------
// Final head: out = relu(h_T @ W_d + b_d), O = 1.
// ---------------------------------------------------------------------------
__global__ void rnn_dense_kernel(const __bf16* __restrict__ h,  // [B,H] bf16
                                 const float* __restrict__ Wd,  // [H,1]
                                 const float* __restrict__ bd,  // [1]
                                 float* __restrict__ out) {     // [B,1]
    const int b = blockIdx.x * blockDim.x + threadIdx.x;
    if (b >= BB) return;
    float s = bd[0];
#pragma unroll 8
    for (int k = 0; k < HH; ++k) s += (float)h[(size_t)b * HH + k] * Wd[k];
    out[b] = s > 0.0f ? s : 0.0f;
}

extern "C" void kernel_launch(void* const* d_in, const int* in_sizes, int n_in,
                              void* d_out, int out_size, void* d_ws, size_t ws_size,
                              hipStream_t stream) {
    const float* x    = (const float*)d_in[0];
    const float* Wx   = (const float*)d_in[1];
    const float* Wh   = (const float*)d_in[2];
    const float* brnn = (const float*)d_in[3];
    const float* Wd   = (const float*)d_in[4];
    const float* bd   = (const float*)d_in[5];
    float* out        = (float*)d_out;

    // workspace layout
    char* ws = (char*)d_ws;
    unsigned int* bar = (unsigned int*)ws;                           // 256 B slot
    __bf16* WTx = (__bf16*)(ws + 256);                               // 64 KB
    __bf16* WTh = (__bf16*)(ws + 256 + 64 * 1024);                   // 512 KB
    __bf16* hb0 = (__bf16*)(ws + 256 + 64 * 1024 + 512 * 1024);      // 2x512 KB ping-pong
    __bf16* hb1 = hb0 + (size_t)BB * HH;

    rnn_setup_kernel<<<(HH * HH + 255) / 256, 256, 0, stream>>>(
        Wx, Wh, WTx, WTh, hb0, bar);

    rnn_step_kernel<<<64, 128, 0, stream>>>(x, brnn, WTx, WTh, hb0, bar);

    // T = 512 (even): last write lands in hb0 ((t+1)&1 with t=511 -> 0)
    rnn_dense_kernel<<<(BB + 127) / 128, 128, 0, stream>>>(hb0, Wd, bd, out);
    (void)hb1; (void)in_sizes; (void)n_in; (void)out_size; (void)ws_size;
}
